// HyperbolicAttH_69260642615721
// MI455X (gfx1250) — matrix operand order repaired and checked by reading, not compiled, not run on hardware
//
#include <hip/hip_runtime.h>
#include <hip/hip_bf16.h>

typedef __attribute__((ext_vector_type(16))) __bf16 v16bf;
typedef __attribute__((ext_vector_type(8)))  float  v8f;
typedef unsigned short ushort_t;

#define EPSF   1e-6f
#define BQ     1024
#define NCAND  40000
#define DDIM   512
#define NCHUNK 40          // ceil(40000 / 1024)
#define ROWS   16
#define NTILE  (NCAND / 16)   // 2500

__device__ __forceinline__ v8f wmma_bf16(v16bf a, v16bf b, v8f c) {
  // (neg_a, A, neg_b, B, c_mod, C, reuse_a, reuse_b)
  return __builtin_amdgcn_wmma_f32_16x16x32_bf16(false, a, false, b, (short)0, c, false, false);
}

__device__ __forceinline__ unsigned short bf_trunc_bits(float f) {
  return (unsigned short)(__float_as_uint(f) >> 16);
}

// ---------------- prep: y_sq[n] = sum_k cand[n][k]^2 (one wave per row) ----------------
__global__ __launch_bounds__(256) void prep_ysq(const float* __restrict__ cand,
                                                float* __restrict__ ysq) {
  const int wave = threadIdx.x >> 5, lane = threadIdx.x & 31;
  const int n = blockIdx.x * 8 + wave;
  if (n >= NCAND) return;
  const float* row = cand + (size_t)n * DDIM;
  float s = 0.f;
  #pragma unroll
  for (int i = 0; i < 16; ++i) { float f = row[lane + (i << 5)]; s = fmaf(f, f, s); }
  s += __shfl_xor(s, 16, 32);
  s += __shfl_xor(s, 8, 32);
  s += __shfl_xor(s, 4, 32);
  s += __shfl_xor(s, 2, 32);
  s += __shfl_xor(s, 1, 32);
  if (lane == 0) ysq[n] = s;
}

// ---- prep: candidates fp32 -> bf16 hi/lo, pre-swizzled into WMMA B-fragment layout ----
// layout: [tile][chunk][lane][16 bf16]; lane<16 -> N=lane, K=e(0..15); lane>=16 -> K=16+e
__global__ __launch_bounds__(256) void prep_swz(const float* __restrict__ cand,
                                                ushort_t* __restrict__ bhi,
                                                ushort_t* __restrict__ blo) {
  const int t    = blockIdx.x;
  const int lane = threadIdx.x & 31;
  const int c0   = threadIdx.x >> 5;      // 0..7
  const int nl   = lane & 15, hf = lane >> 4;
  const float* crow = cand + (size_t)(t * 16 + nl) * DDIM + (hf << 4);
  #pragma unroll
  for (int cc = 0; cc < 2; ++cc) {
    int c = c0 + (cc << 3);
    const float4* p = (const float4*)(crow + (c << 5));
    float4 q0 = p[0], q1 = p[1], q2 = p[2], q3 = p[3];
    float fs[16] = {q0.x, q0.y, q0.z, q0.w, q1.x, q1.y, q1.z, q1.w,
                    q2.x, q2.y, q2.z, q2.w, q3.x, q3.y, q3.z, q3.w};
    union { ushort_t u[16]; uint4 q[2]; } H, L;
    #pragma unroll
    for (int e = 0; e < 16; ++e) {
      float f = fs[e];
      unsigned u = __float_as_uint(f);
      float fh = __uint_as_float(u & 0xffff0000u);
      H.u[e] = (ushort_t)(u >> 16);
      L.u[e] = bf_trunc_bits(f - fh);
    }
    size_t ob = ((size_t)(t * 16 + c) * 32 + lane) * 16;   // 32B-aligned
    uint4* dh = (uint4*)(bhi + ob);
    uint4* dl = (uint4*)(blo + ob);
    dh[0] = H.q[0]; dh[1] = H.q[1];
    dl[0] = L.q[0]; dl[1] = L.q[1];
  }
}

// ---------------------------------- fused main kernel ----------------------------------
template <bool PRE>
__global__ __launch_bounds__(256) void hyper_ce_main(
    const float* __restrict__ query,
    const float* __restrict__ cand,
    const float* __restrict__ cbias,
    const float* __restrict__ curv,
    const int*   __restrict__ target,
    const ushort_t* __restrict__ wbhi,
    const ushort_t* __restrict__ wblo,
    const float* __restrict__ wysq,
    float* __restrict__ pmax,
    float* __restrict__ psum,
    float* __restrict__ ptgt)
{
  __shared__ __align__(32) unsigned short ldsAhi[16 * 32 * 16]; // [chunk][lane][16]
  __shared__ __align__(32) unsigned short ldsAlo[16 * 32 * 16];
  __shared__ float s_xsq[ROWS], s_cq[ROWS], s_sqc[ROWS], s_inv[ROWS], s_mxn[ROWS];
  __shared__ float s_2bc[ROWS], s_bc2[ROWS], s_cqxsq[ROWS], s_2cq[ROWS];
  __shared__ int   s_tgt[ROWS];
  __shared__ float s_cmax[8][ROWS], s_csum[8][ROWS], s_ctgt[8][ROWS];

  const int tid    = threadIdx.x;
  const int lane   = tid & 31;
  const int wave   = tid >> 5;
  const int hf     = lane >> 4;
  const int nl     = lane & 15;
  const int b0     = blockIdx.y * ROWS;
  const int chunkx = blockIdx.x;

  if (tid < ROWS) {
    s_cq[tid]  = curv[b0 + tid];
    s_tgt[tid] = target[b0 + tid];
  }

  // ---- stage 16 query rows into LDS (WMMA A-fragment layout, bf16 hi/lo); x_sq ----
  {
    const int m  = tid >> 4;
    const int kb = tid & 15;
    const float* qrow = query + (size_t)(b0 + m) * DDIM;
    float xp = 0.f;
    #pragma unroll
    for (int j = 0; j < 32; ++j) {
      int k = kb + (j << 4);
      float f = qrow[k];
      xp = fmaf(f, f, xp);
      unsigned u = __float_as_uint(f);
      float fh = __uint_as_float(u & 0xffff0000u);
      float fl = f - fh;
      int c = k >> 5, kl = k & 31;
      int ln, e;
      ln = m + ((kl >> 4) << 4);
      e  = kl & 15;
      int idx = (((c << 5) + ln) << 4) + e;
      ldsAhi[idx] = (unsigned short)(u >> 16);
      ldsAlo[idx] = bf_trunc_bits(fl);
    }
    xp += __shfl_xor(xp, 1, 32);
    xp += __shfl_xor(xp, 2, 32);
    xp += __shfl_xor(xp, 4, 32);
    xp += __shfl_xor(xp, 8, 32);
    if ((tid & 15) == 0) s_xsq[m] = xp;
  }
  __syncthreads();

  // ---- per-row constants (cold path, full precision) ----
  if (tid < ROWS) {
    float cq  = s_cq[tid];
    float xsq = s_xsq[tid];
    float sqc = sqrtf(cq + EPSF);
    float inv = 1.f / (sqc + EPSF);
    float Bc  = 1.f - cq * xsq;
    s_sqc[tid]   = sqc;
    s_inv[tid]   = inv;
    s_mxn[tid]   = inv - EPSF;
    s_2bc[tid]   = 2.f * Bc;
    s_bc2[tid]   = Bc * Bc;
    s_cqxsq[tid] = cq * xsq;
    s_2cq[tid]   = 2.f * cq;
  }
  __syncthreads();

  float rmax[8], rsum[8], rtgt[8];
  #pragma unroll
  for (int r = 0; r < 8; ++r) { rmax[r] = -INFINITY; rsum[r] = 0.f; rtgt[r] = -INFINITY; }

  for (int it = 0; it < 8; ++it) {
    int tile = chunkx * 64 + it * 8 + wave;
    int n0 = tile << 4;
    if (n0 >= NCAND) continue;               // wave-uniform (40000 % 16 == 0)

    v8f acc = {};
    float ysq;
    const int n = n0 + nl;

    if (PRE) {
      const size_t bbase = (size_t)tile * 8192 + (size_t)lane * 16;
      #pragma unroll
      for (int c = 0; c < 16; ++c) {
        v16bf bh = *(const v16bf*)&wbhi[bbase + (size_t)c * 512];
        v16bf bl = *(const v16bf*)&wblo[bbase + (size_t)c * 512];
        int abase = ((c << 5) + lane) << 4;
        v16bf ah = *(const v16bf*)&ldsAhi[abase];
        v16bf al = *(const v16bf*)&ldsAlo[abase];
        acc = wmma_bf16(ah, bh, acc);
        acc = wmma_bf16(ah, bl, acc);
        acc = wmma_bf16(al, bh, acc);
      }
      ysq = wysq[n];
    } else {
      const float* crow = cand + (size_t)n * DDIM + (hf << 4);
      float ysqh = 0.f;
      #pragma unroll
      for (int c = 0; c < 16; ++c) {
        const float4* p = (const float4*)(crow + (c << 5));
        float4 q0 = p[0], q1 = p[1], q2 = p[2], q3 = p[3];
        float fs[16] = {q0.x, q0.y, q0.z, q0.w, q1.x, q1.y, q1.z, q1.w,
                        q2.x, q2.y, q2.z, q2.w, q3.x, q3.y, q3.z, q3.w};
        v16bf bh, bl;
        #pragma unroll
        for (int e = 0; e < 16; ++e) {
          float f = fs[e];
          ysqh = fmaf(f, f, ysqh);
          unsigned u = __float_as_uint(f);
          float fh = __uint_as_float(u & 0xffff0000u);
          bh[e] = __builtin_bit_cast(__bf16, (unsigned short)(u >> 16));
          bl[e] = __builtin_bit_cast(__bf16, bf_trunc_bits(f - fh));
        }
        int abase = ((c << 5) + lane) << 4;
        v16bf ah = *(const v16bf*)&ldsAhi[abase];
        v16bf al = *(const v16bf*)&ldsAlo[abase];
        acc = wmma_bf16(ah, bh, acc);
        acc = wmma_bf16(ah, bl, acc);
        acc = wmma_bf16(al, bh, acc);
      }
      ysq = ysqh + __shfl_xor(ysqh, 16, 32);
    }

    float bias = cbias[n];

    #pragma unroll
    for (int r = 0; r < 8; ++r) {
      int m = r + (hf << 3);
      float xy  = acc[r];
      float t2  = s_cq[m] * ysq;                      // c*y^2
      float tx  = s_2cq[m] * xy;                      // 2*c*xy
      float A   = 1.f + t2 - tx;
      float den = fmaf(s_cqxsq[m], t2, 1.f) - tx;     // 1 - 2c*xy + c^2*x^2*y^2
      float num = fmaf(A, fmaf(A, s_xsq[m], -s_2bc[m] * xy), s_bc2[m] * ysq);
      float diff = __fdividef(sqrtf(fmaxf(num, 0.f)), den + EPSF);
      diff = fmaxf(diff, EPSF);
      diff = fminf(diff, s_mxn[m]);
      float arg = fminf(s_sqc[m] * diff, 1.f - EPSF);
      float dist = s_inv[m] * __logf(__fdividef(1.f + arg, 1.f - arg));
      float logit = bias - dist;

      float nm = fmaxf(rmax[r], logit);
      rsum[r] = rsum[r] * __expf(rmax[r] - nm) + __expf(logit - nm);
      rmax[r] = nm;
      if (n == s_tgt[m]) rtgt[r] = logit;
    }
  }

  // ---- per-wave cross-lane (16) logsumexp / target reduce ----
  bool waveEmpty = (((chunkx * 64 + wave) << 4) >= NCAND);   // iter-0 tile; uniform
  if (!waveEmpty) {
    #pragma unroll
    for (int r = 0; r < 8; ++r) {
      #pragma unroll
      for (int msk = 1; msk < 16; msk <<= 1) {
        float om = __shfl_xor(rmax[r], msk, 32);
        float os = __shfl_xor(rsum[r], msk, 32);
        float ot = __shfl_xor(rtgt[r], msk, 32);
        float nm = fmaxf(rmax[r], om);
        rsum[r] = rsum[r] * __expf(rmax[r] - nm) + os * __expf(om - nm);
        rmax[r] = nm;
        rtgt[r] = fmaxf(rtgt[r], ot);
      }
    }
  }
  if ((lane & 15) == 0) {
    #pragma unroll
    for (int r = 0; r < 8; ++r) {
      int m = r + (hf << 3);
      s_cmax[wave][m] = rmax[r];
      s_csum[wave][m] = rsum[r];
      s_ctgt[wave][m] = rtgt[r];
    }
  }
  __syncthreads();

  if (tid < ROWS) {
    float m0 = s_cmax[0][tid], s0 = s_csum[0][tid], t0 = s_ctgt[0][tid];
    #pragma unroll
    for (int w = 1; w < 8; ++w) {
      float mw = s_cmax[w][tid], sw = s_csum[w][tid], tw = s_ctgt[w][tid];
      float nm = fmaxf(m0, mw);
      s0 = s0 * __expf(m0 - nm) + sw * __expf(mw - nm);
      m0 = nm;
      t0 = fmaxf(t0, tw);
    }
    int row = b0 + tid;
    pmax[row * NCHUNK + chunkx] = m0;
    psum[row * NCHUNK + chunkx] = s0;
    ptgt[row * NCHUNK + chunkx] = t0;
  }
}

// -------------------------------- final reduction --------------------------------
__global__ __launch_bounds__(1024) void hyper_ce_final(
    const float* __restrict__ pmax,
    const float* __restrict__ psum,
    const float* __restrict__ ptgt,
    float* __restrict__ out)
{
  __shared__ float red[BQ];
  int b = threadIdx.x;
  float m = pmax[b * NCHUNK], s = psum[b * NCHUNK], tg = ptgt[b * NCHUNK];
  for (int c = 1; c < NCHUNK; ++c) {
    float mw = pmax[b * NCHUNK + c], sw = psum[b * NCHUNK + c];
    float nm = fmaxf(m, mw);
    s = s * expf(m - nm) + sw * expf(mw - nm);
    m = nm;
    tg = fmaxf(tg, ptgt[b * NCHUNK + c]);
  }
  red[b] = (m + logf(s)) - tg;
  __syncthreads();
  for (int off = 512; off > 0; off >>= 1) {
    if (b < off) red[b] += red[b + off];
    __syncthreads();
  }
  if (b == 0) out[0] = red[0] * (1.0f / (float)BQ);
}

extern "C" void kernel_launch(void* const* d_in, const int* in_sizes, int n_in,
                              void* d_out, int out_size, void* d_ws, size_t ws_size,
                              hipStream_t stream) {
  const float* query = (const float*)d_in[0];
  const float* cand  = (const float*)d_in[1];
  const float* cbias = (const float*)d_in[2];
  const float* curv  = (const float*)d_in[3];
  const int*   tgt   = (const int*)d_in[4];
  float* out  = (float*)d_out;

  const size_t P = (size_t)BQ * NCHUNK;                  // 40960 per partial array
  float*    pmax = (float*)d_ws;
  float*    psum = pmax + P;
  float*    ptgt = psum + P;
  float*    ysq  = ptgt + P;                             // NCAND floats
  ushort_t* bhi  = (ushort_t*)(ysq + NCAND);             // 32B-aligned offset
  ushort_t* blo  = bhi + (size_t)NTILE * 8192;

  const size_t need = (3 * P + NCAND) * sizeof(float)
                    + 2 * (size_t)NTILE * 8192 * sizeof(ushort_t);   // 82,571,520 B
  const bool pre = (ws_size >= need);

  dim3 grid(NCHUNK, BQ / ROWS);
  if (pre) {
    prep_ysq<<<NCAND / 8, 256, 0, stream>>>(cand, ysq);
    prep_swz<<<NTILE, 256, 0, stream>>>(cand, bhi, blo);
    hyper_ce_main<true><<<grid, 256, 0, stream>>>(
        query, cand, cbias, curv, tgt, bhi, blo, ysq, pmax, psum, ptgt);
  } else {
    hyper_ce_main<false><<<grid, 256, 0, stream>>>(
        query, cand, cbias, curv, tgt, bhi, blo, ysq, pmax, psum, ptgt);
  }
  hyper_ce_final<<<1, BQ, 0, stream>>>(pmax, psum, ptgt, out);
}
